// MultiHeadSelfAttention_60971355734426
// MI455X (gfx1250) — compile-verified, matrix operand order checked by reading, not run on hardware
//
#include <hip/hip_runtime.h>

// ---------------------------------------------------------------------------
// MI455X (gfx1250, wave32) multi-head self-attention forward.
// Matrix math: v_wmma_f32_16x16x32_bf16 (f32 accumulate).
// Data movement: global_load_async_to_lds_b128 (GEMM tiles, double-buffered)
//                tensor_load_to_lds / TDM     (attention K/V tiles, db-buffered)
// Fixed sizes: B=2, T=2048, D=1024, H=16, DH=64.
// ---------------------------------------------------------------------------

typedef unsigned int   u32;
typedef unsigned short u16;
typedef unsigned long long u64;

typedef __bf16 v16bf __attribute__((ext_vector_type(16)));
typedef float  v8f   __attribute__((ext_vector_type(8)));
typedef unsigned int v4u __attribute__((ext_vector_type(4)));
typedef int          v8i __attribute__((ext_vector_type(8)));
typedef int          v4i __attribute__((ext_vector_type(4)));

union FragBF { v16bf v; u32 u[8]; };

__device__ __forceinline__ u16 f2bf(float f) {
  u32 u = __float_as_uint(f);
  return (u16)((u + 0x7FFFu + ((u >> 16) & 1u)) >> 16);  // round-to-nearest-even
}

// 16-bit A-matrix K index for fragment dword i (0..7), lane group g (= lane>>4).
__device__ __forceinline__ int kb16(int i, int g) {
  return (i < 4 ? (i << 1) : (16 + ((i - 4) << 1))) + (g << 3);
}

#define WMMA_BF16(a, b, c) \
  __builtin_amdgcn_wmma_f32_16x16x32_bf16(false, (a), false, (b), (short)0, (c), false, false)

// Per-lane async copy: 16 B global -> LDS, tracked by ASYNCcnt (no VGPR staging).
__device__ __forceinline__ void async_b128(u32 lds_off, const u16* g) {
  asm volatile("global_load_async_to_lds_b128 %0, %1, off"
               :: "v"(lds_off), "v"(g) : "memory");
}
__device__ __forceinline__ void wait_async0() {
  asm volatile("s_wait_asynccnt 0" ::: "memory");
}

// TDM: DMA one 2-D bf16 tile (64 rows x 64 elems, row stride in elements) into
// LDS, padding +4 DWORDs after every row (32 DWORDs) -> LDS row stride 72 u16.
__device__ __forceinline__ void tdm_load_64x64_bf16(u32 lds_off, const u16* gbase,
                                                    u32 stride_elems) {
  u64 ga = (u64)(uintptr_t)gbase;
  v4u g0;
  g0[0] = 1u;                                 // count=1, gather off
  g0[1] = lds_off;                            // LDS byte address
  g0[2] = (u32)ga;                            // global addr [31:0]
  g0[3] = (u32)(ga >> 32) | 0x80000000u;      // global addr [56:32] | type=2
  v8i g1;
  g1[0] = (int)((1u << 16) |                  // data_size = 2 B
                (1u << 20) |                  // pad_enable
                (4u << 22) |                  // pad_interval: every 32 DWORDs
                (3u << 25));                  // pad_amount: 4 DWORDs
  g1[1] = (int)(64u << 16);                   // tensor_dim0 = 64
  g1[2] = (int)(64u << 16);                   // tensor_dim1 = 64
  g1[3] = (int)(64u << 16);                   // tile_dim0 = 64
  g1[4] = (int)64u;                           // tile_dim1 = 64, tile_dim2 = 0
  g1[5] = (int)stride_elems;                  // tensor_dim0_stride
  g1[6] = 0;
  g1[7] = 0;                                  // tensor_dim1_stride = 0 (2-D)
  v4i z4; z4[0] = 0; z4[1] = 0; z4[2] = 0; z4[3] = 0;
  v8i z8; z8[0] = 0; z8[1] = 0; z8[2] = 0; z8[3] = 0;
  z8[4] = 0; z8[5] = 0; z8[6] = 0; z8[7] = 0;
  // 6-arg form (clang-23 / therock-10.0 headers): (g0, g1, g2, g3, g4, cpol)
  __builtin_amdgcn_tensor_load_to_lds(g0, g1, z4, z4, z8, 0);
}

// ---------------------------------------------------------------------------
// Kernel 1a: f32 -> bf16 elementwise (for x)
// ---------------------------------------------------------------------------
__global__ __launch_bounds__(256) void cast_f32_bf16_k(const float* __restrict__ in,
                                                       u16* __restrict__ out, int n) {
  int i = blockIdx.x * 256 + threadIdx.x;
  if (i < n) out[i] = f2bf(in[i]);
}

// ---------------------------------------------------------------------------
// Kernel 1b: f32 [1024][1024] -> bf16 transposed (LDS-tiled 32x32)
// ---------------------------------------------------------------------------
__global__ __launch_bounds__(256) void transpose_f32_bf16_k(const float* __restrict__ in,
                                                            u16* __restrict__ out) {
  __shared__ float tile[32][33];
  int x = blockIdx.x * 32 + threadIdx.x;
  int y = blockIdx.y * 32 + threadIdx.y;
#pragma unroll
  for (int k = 0; k < 32; k += 8)
    tile[threadIdx.y + k][threadIdx.x] = in[(size_t)(y + k) << 10 | x];
  __syncthreads();
  int xo = blockIdx.y * 32 + threadIdx.x;
  int yo = blockIdx.x * 32 + threadIdx.y;
#pragma unroll
  for (int k = 0; k < 32; k += 8)
    out[(size_t)(yo + k) << 10 | xo] = f2bf(tile[threadIdx.x][threadIdx.y + k]);
}

// ---------------------------------------------------------------------------
// Kernel 2/5: bf16 WMMA GEMM, C = A[4096x1024] * Bt[1024x1024]^T.
// 128x128 block tile, 8 waves (2x4), wave tile 64x32 (4x2 WMMA), K-step 32.
// Double-buffered LDS filled by async-to-LDS loads (ASYNCcnt), 1 barrier/step.
// mode 0: bf16 out, [B,H,T,DH] (Q/K projections, scaled)
// mode 1: bf16 out, [B,H,DH,T] (V projection, transposed)
// mode 2: f32  out, row-major [4096][1024] (final projection)
// ---------------------------------------------------------------------------
__global__ __launch_bounds__(256) void wmma_gemm_bf16_k(
    const u16* __restrict__ A, const u16* __restrict__ Bt, void* __restrict__ C,
    int mode, float scale) {
  constexpr int Kd  = 1024;
  constexpr int LDA = 40;  // 80 B rows: 16B-aligned b128, conflict-free frag reads
  __shared__ alignas(16) u16 As[2][128][LDA];
  __shared__ alignas(16) u16 Bs[2][128][LDA];

  const int tid  = threadIdx.x;
  const int lane = tid & 31, wid = tid >> 5;
  const int g = lane >> 4, lm = lane & 15;
  const int waveM = (wid >> 2) * 64;
  const int waveN = (wid & 3) * 32;
  const int rowBase = blockIdx.y * 128;
  const int colBase = blockIdx.x * 128;

  v8f acc[4][2];
#pragma unroll
  for (int mt = 0; mt < 4; ++mt)
#pragma unroll
    for (int nt = 0; nt < 2; ++nt)
#pragma unroll
      for (int e = 0; e < 8; ++e) acc[mt][nt][e] = 0.0f;

  // Issue one 128x32 A tile + 128x32 Bt tile into LDS buffer (ki&1).
  auto issue = [&](int ki) {
    const int buf = ki & 1, k0 = ki << 5;
#pragma unroll
    for (int it = 0; it < 2; ++it) {
      int pos = tid + it * 256;
      int r = pos >> 2, c = (pos & 3) << 3;
      async_b128((u32)(uintptr_t)&As[buf][r][c], A  + ((size_t)(rowBase + r) << 10) + k0 + c);
      async_b128((u32)(uintptr_t)&Bs[buf][r][c], Bt + ((size_t)(colBase + r) << 10) + k0 + c);
    }
  };

  constexpr int nK = Kd / 32;
  issue(0);
  for (int ki = 0; ki < nK; ++ki) {
    wait_async0();        // own async ops done ...
    __syncthreads();      // ... so after barrier the whole tile is resident
    if (ki + 1 < nK) issue(ki + 1);  // prefetch overlaps the 8 WMMAs below
    const int buf = ki & 1;

    FragBF bfr[2];
#pragma unroll
    for (int nt = 0; nt < 2; ++nt)
#pragma unroll
      for (int i = 0; i < 8; ++i)
        bfr[nt].u[i] = *(const u32*)&Bs[buf][waveN + nt * 16 + lm][kb16(i, g)];

#pragma unroll
    for (int mt = 0; mt < 4; ++mt) {
      FragBF afr;
#pragma unroll
      for (int i = 0; i < 8; ++i)
        afr.u[i] = *(const u32*)&As[buf][waveM + mt * 16 + lm][kb16(i, g)];
#pragma unroll
      for (int nt = 0; nt < 2; ++nt)
        acc[mt][nt] = WMMA_BF16(afr.v, bfr[nt].v, acc[mt][nt]);
    }
  }

  // Epilogue. C-layout: dword r, lane -> (m = r + 8g, n = lane&15).
#pragma unroll
  for (int mt = 0; mt < 4; ++mt)
#pragma unroll
    for (int nt = 0; nt < 2; ++nt)
#pragma unroll
      for (int r = 0; r < 8; ++r) {
        int m = rowBase + waveM + mt * 16 + r + 8 * g;
        int n = colBase + waveN + nt * 16 + lm;
        float v = acc[mt][nt][r] * scale;
        if (mode == 2) {
          ((float*)C)[(size_t)m << 10 | n] = v;
        } else {
          int b = m >> 11, t = m & 2047;     // T = 2048
          int h = n >> 6,  dh = n & 63;      // DH = 64
          if (mode == 0)  // [B,H,T,DH]
            ((u16*)C)[((size_t)((b * 16 + h) * 2048 + t) << 6) | dh] = f2bf(v);
          else            // [B,H,DH,T]
            ((u16*)C)[((size_t)((b * 16 + h) * 64 + dh) << 11) | t] = f2bf(v);
        }
      }
}

// ---------------------------------------------------------------------------
// Kernel 3: flash-style causal attention, DH=64, T=2048, H=16.
// Grid: (T/64, B*H). 128 threads = 4 waves; wave w owns q-rows [qbase+16w, +16).
// K/V tiles DMA'd by the Tensor Data Mover (wave 0 issues, TENSORcnt),
// double-buffered so the next tile streams in during the current tile's math.
// Q pre-scaled by 1/sqrt(DH) at projection time.
// ---------------------------------------------------------------------------
__global__ __launch_bounds__(128) void flash_attn_causal_k(
    const u16* __restrict__ Q, const u16* __restrict__ K, const u16* __restrict__ Vt,
    u16* __restrict__ O) {
  constexpr int T = 2048;
  constexpr int LDS_W = 72;  // TDM pad: 64 u16 row + 4 DWORDs = 72 u16 stride
  __shared__ alignas(16) u16 Ks[2][64][LDS_W];
  __shared__ alignas(16) u16 Vs[2][64][LDS_W];   // Vt tile: [dh][key]
  __shared__ alignas(16) u16 Ps[4][16][LDS_W];

  const int tid  = threadIdx.x;
  const int lane = tid & 31, wid = tid >> 5;
  const int g = lane >> 4, lm = lane & 15;
  const int bh = blockIdx.y;
  const int qtile = blockIdx.x;
  const int qbase = qtile * 64;
  const int qrow = qbase + wid * 16;

  // Q A-fragments (rows qrow+lm), two K-steps over DH=64.
  FragBF q0, q1;
  const u16* Qp = Q + (((size_t)bh * T + qrow + lm) << 6);
#pragma unroll
  for (int i = 0; i < 8; ++i) {
    int kb = kb16(i, g);
    q0.u[i] = *(const u32*)(Qp + kb);
    q1.u[i] = *(const u32*)(Qp + 32 + kb);
  }

  v8f oacc[4];
#pragma unroll
  for (int dt = 0; dt < 4; ++dt)
#pragma unroll
    for (int e = 0; e < 8; ++e) oacc[dt][e] = 0.0f;
  float mstate[8], lstate[8];
#pragma unroll
  for (int r = 0; r < 8; ++r) { mstate[r] = -3.0e38f; lstate[r] = 0.0f; }

  auto issue_tiles = [&](int j) {  // TDM DMA of K and V^T tiles for key tile j
    const int jbase = j * 64, buf = j & 1;
    tdm_load_64x64_bf16((u32)(uintptr_t)&Ks[buf][0][0],
                        K + (((size_t)bh * T + jbase) << 6), 64);
    tdm_load_64x64_bf16((u32)(uintptr_t)&Vs[buf][0][0],
                        Vt + (((size_t)bh << 6) << 11) + jbase, T);
  };

  if (wid == 0) issue_tiles(0);

  for (int j = 0; j <= qtile; ++j) {  // causal: lower-triangular key tiles only
    if (wid == 0) __builtin_amdgcn_s_wait_tensorcnt(0);
    __syncthreads();                              // tile j resident for all waves
    if (wid == 0 && j < qtile) issue_tiles(j + 1);  // DMA next tile during math
    const int buf = j & 1;
    const int jbase = j * 64;

    // S = Q * K^T  (Bt layout = K tile: lane n -> key, k -> dh)
    v8f s[4];
#pragma unroll
    for (int nt = 0; nt < 4; ++nt) {
      v8f sa;
#pragma unroll
      for (int e = 0; e < 8; ++e) sa[e] = 0.0f;
      FragBF b0, b1;
#pragma unroll
      for (int i = 0; i < 8; ++i) {
        int kb = kb16(i, g);
        b0.u[i] = *(const u32*)&Ks[buf][nt * 16 + lm][kb];
        b1.u[i] = *(const u32*)&Ks[buf][nt * 16 + lm][32 + kb];
      }
      sa = WMMA_BF16(q0.v, b0.v, sa);
      sa = WMMA_BF16(q1.v, b1.v, sa);
      s[nt] = sa;
    }

    if (j == qtile) {  // diagonal tile: causal mask as elementwise select
#pragma unroll
      for (int nt = 0; nt < 4; ++nt)
#pragma unroll
        for (int r = 0; r < 8; ++r) {
          int ki = jbase + nt * 16 + lm;
          int qi = qrow + r + 8 * g;
          s[nt][r] = (ki > qi) ? -3.0e38f : s[nt][r];
        }
    }

    // Online softmax. Row (r+8g) lives in lane-group g across 16 lanes, reg r.
#pragma unroll
    for (int r = 0; r < 8; ++r) {
      float mx = fmaxf(fmaxf(s[0][r], s[1][r]), fmaxf(s[2][r], s[3][r]));
#pragma unroll
      for (int d = 1; d < 16; d <<= 1) mx = fmaxf(mx, __shfl_xor(mx, d, 32));
      float mnew = fmaxf(mstate[r], mx);
      float alpha = __expf(mstate[r] - mnew);
      mstate[r] = mnew;
      float rs = 0.0f;
#pragma unroll
      for (int nt = 0; nt < 4; ++nt) {
        float pv = __expf(s[nt][r] - mnew);
        s[nt][r] = pv;
        rs += pv;
      }
#pragma unroll
      for (int d = 1; d < 16; d <<= 1) rs += __shfl_xor(rs, d, 32);
      lstate[r] = lstate[r] * alpha + rs;
#pragma unroll
      for (int dt = 0; dt < 4; ++dt) oacc[dt][r] *= alpha;
    }

    // Re-layout P (C-layout -> A-layout) through per-wave LDS.
#pragma unroll
    for (int nt = 0; nt < 4; ++nt)
#pragma unroll
      for (int r = 0; r < 8; ++r)
        Ps[wid][r + 8 * g][nt * 16 + lm] = f2bf(s[nt][r]);
    __syncthreads();

    // O += P * V  (Bt layout = Vt tile: lane n -> dh, k -> key)
    FragBF p0, p1;
#pragma unroll
    for (int i = 0; i < 8; ++i) {
      int kb = kb16(i, g);
      p0.u[i] = *(const u32*)&Ps[wid][lm][kb];
      p1.u[i] = *(const u32*)&Ps[wid][lm][32 + kb];
    }
#pragma unroll
    for (int dt = 0; dt < 4; ++dt) {
      FragBF b0, b1;
#pragma unroll
      for (int i = 0; i < 8; ++i) {
        int kb = kb16(i, g);
        b0.u[i] = *(const u32*)&Vs[buf][dt * 16 + lm][kb];
        b1.u[i] = *(const u32*)&Vs[buf][dt * 16 + lm][32 + kb];
      }
      oacc[dt] = WMMA_BF16(p0.v, b0.v, oacc[dt]);
      oacc[dt] = WMMA_BF16(p1.v, b1.v, oacc[dt]);
    }
    // no trailing barrier: next iteration's wait+barrier orders buffer reuse
  }

  // Normalize and store O as bf16 [B, T, D] (h-interleaved => A-layout for Wo).
  const int b = bh >> 4, h = bh & 15;
#pragma unroll
  for (int dt = 0; dt < 4; ++dt)
#pragma unroll
    for (int r = 0; r < 8; ++r) {
      int t = qrow + r + 8 * g;
      int dh = dt * 16 + lm;
      O[((size_t)(b * T + t) << 10) | (h * 64 + dh)] = f2bf(oacc[dt][r] / lstate[r]);
    }
}

// ---------------------------------------------------------------------------
// Host-side launch
// ---------------------------------------------------------------------------
extern "C" void kernel_launch(void* const* d_in, const int* in_sizes, int n_in,
                              void* d_out, int out_size, void* d_ws, size_t ws_size,
                              hipStream_t stream) {
  (void)in_sizes; (void)n_in; (void)out_size; (void)ws_size;
  const int B = 2, T = 2048, D = 1024;
  const int M = B * T;

  const float* x  = (const float*)d_in[0];
  // d_in[1] = mask (bool) — causality computed analytically.
  const float* Wq = (const float*)d_in[2];
  const float* Wk = (const float*)d_in[3];
  const float* Wv = (const float*)d_in[4];
  const float* Wo = (const float*)d_in[5];

  char* base = (char*)d_ws;
  size_t off = 0;
  auto carve = [&](size_t bytes) -> void* {
    void* p = base + off;
    off += (bytes + 255) & ~(size_t)255;
    return p;
  };
  u16* xb  = (u16*)carve((size_t)M * D * 2);
  u16* WqT = (u16*)carve((size_t)D * D * 2);
  u16* WkT = (u16*)carve((size_t)D * D * 2);
  u16* WvT = (u16*)carve((size_t)D * D * 2);
  u16* WoT = (u16*)carve((size_t)D * D * 2);
  u16* Qb  = (u16*)carve((size_t)M * D * 2);   // [B,H,T,DH], pre-scaled
  u16* Kb  = (u16*)carve((size_t)M * D * 2);   // [B,H,T,DH]
  u16* Vtb = (u16*)carve((size_t)M * D * 2);   // [B,H,DH,T]
  u16* Ob  = (u16*)carve((size_t)M * D * 2);   // [B,T,D]

  {
    int n = M * D;
    cast_f32_bf16_k<<<n / 256, 256, 0, stream>>>(x, xb, n);
    dim3 g(D / 32, D / 32), blk(32, 8);
    transpose_f32_bf16_k<<<g, blk, 0, stream>>>(Wq, WqT);
    transpose_f32_bf16_k<<<g, blk, 0, stream>>>(Wk, WkT);
    transpose_f32_bf16_k<<<g, blk, 0, stream>>>(Wv, WvT);
    transpose_f32_bf16_k<<<g, blk, 0, stream>>>(Wo, WoT);
  }
  {
    dim3 g(D / 128, M / 128), blk(256);
    wmma_gemm_bf16_k<<<g, blk, 0, stream>>>(xb, WqT, Qb, 0, 0.125f);
    wmma_gemm_bf16_k<<<g, blk, 0, stream>>>(xb, WkT, Kb, 0, 1.0f);
    wmma_gemm_bf16_k<<<g, blk, 0, stream>>>(xb, WvT, Vtb, 1, 1.0f);
  }
  {
    dim3 g(T / 64, B * 16), blk(128);
    flash_attn_causal_k<<<g, blk, 0, stream>>>(Qb, Kb, Vtb, Ob);
  }
  {
    dim3 g(D / 128, M / 128), blk(256);
    wmma_gemm_bf16_k<<<g, blk, 0, stream>>>(Ob, WoT, d_out, 2, 1.0f);
  }
}
